// MergeLayerC_5231270167275
// MI455X (gfx1250) — matrix operand order repaired and checked by reading, not compiled, not run on hardware
//
#include <hip/hip_runtime.h>
#include <hip/hip_bf16.h>

// ---------------------------------------------------------------------------
// MergeLayerC on MI455X (gfx1250, wave32).
// conv0 (15x15, 8->4 ch) is cast as implicit GEMM on v_wmma_f32_16x16x32_f16
// (f16 in, f32 accumulate). Input tiles staged into LDS with the CDNA5 async
// global->LDS path where available. Everything else is bandwidth-bound passes.
// ---------------------------------------------------------------------------

typedef __attribute__((ext_vector_type(16))) _Float16 v16h;
typedef __attribute__((ext_vector_type(8)))  float    v8f;

#define HH 512
#define WW 512
#define HWs (512LL * 512LL)

// --- CDNA5 async global->LDS copy (guarded; falls back to VGPR staging) -----
#if defined(__has_builtin)
#  if __has_builtin(__builtin_amdgcn_global_load_async_to_lds_b32)
#    define HAVE_ASYNC_LDS 1
#  endif
#endif
#ifndef HAVE_ASYNC_LDS
#  define HAVE_ASYNC_LDS 0
#endif

#if HAVE_ASYNC_LDS
typedef __attribute__((address_space(1))) int gint_as;   // global int
typedef __attribute__((address_space(3))) int lint_as;   // LDS int
__device__ __forceinline__ void async_copy_b32(const void* g, void* l)
{
    __builtin_amdgcn_global_load_async_to_lds_b32(
        (gint_as*)(unsigned long long)(uintptr_t)g,
        (lint_as*)(unsigned int)(uintptr_t)l, 0, 0);
}
__device__ __forceinline__ void async_wait0()
{
#  if __has_builtin(__builtin_amdgcn_s_wait_asynccnt)
    __builtin_amdgcn_s_wait_asynccnt(0);
#  else
    asm volatile("s_wait_asynccnt 0x0" ::: "memory");
#  endif
}
#endif

// xor-butterfly add within each 16-lane half via ds_swizzle (wave32)
__device__ __forceinline__ float half_wave_sum(float s)
{
    s += __int_as_float(__builtin_amdgcn_ds_swizzle(__float_as_int(s), 0x041F));
    s += __int_as_float(__builtin_amdgcn_ds_swizzle(__float_as_int(s), 0x081F));
    s += __int_as_float(__builtin_amdgcn_ds_swizzle(__float_as_int(s), 0x101F));
    s += __int_as_float(__builtin_amdgcn_ds_swizzle(__float_as_int(s), 0x201F));
    return s;
}

// ---------------------------------------------------------------------------
// Kernel 1: bilinear warp of x[:,4:8] by flow x[:,8:10], concat with x[:,:4].
// xc layout: (8, 8, 512, 512)
// ---------------------------------------------------------------------------
__global__ void warp_concat_kernel(const float* __restrict__ x,
                                   float* __restrict__ xc,
                                   int total)
{
    int idx = blockIdx.x * blockDim.x + threadIdx.x;
    if (idx >= total) return;
    int xp  = idx % WW;
    int t   = idx / WW;
    int y   = t % HH;
    int b   = t / HH;

    const float* xb = x + (long long)b * 10 * HWs;
    float*       ob = xc + (long long)b * 8 * HWs;
    long long pix = (long long)y * WW + xp;

    #pragma unroll
    for (int c = 0; c < 4; ++c)
        ob[c * HWs + pix] = xb[c * HWs + pix];

    float fx = xb[8 * HWs + pix];
    float fy = xb[9 * HWs + pix];
    float gx = (float)xp + fx;
    float gy = (float)y  + fy;
    float x0f = floorf(gx), y0f = floorf(gy);
    float wx1 = gx - x0f,   wy1 = gy - y0f;

    float acc[4] = {0.f, 0.f, 0.f, 0.f};
    float msk = 0.f;
    #pragma unroll
    for (int dx = 0; dx < 2; ++dx) {
        float wx = dx ? wx1 : (1.f - wx1);
        float xcf = x0f + (float)dx;
        #pragma unroll
        for (int dy = 0; dy < 2; ++dy) {
            float wy = dy ? wy1 : (1.f - wy1);
            float ycf = y0f + (float)dy;
            bool valid = (xcf >= 0.f) && (xcf <= (float)(WW - 1)) &&
                         (ycf >= 0.f) && (ycf <= (float)(HH - 1));
            float w2 = wx * wy * (valid ? 1.f : 0.f);
            int xi = (int)fminf(fmaxf(xcf, 0.f), (float)(WW - 1));
            int yi = (int)fminf(fmaxf(ycf, 0.f), (float)(HH - 1));
            long long gp = (long long)yi * WW + xi;
            #pragma unroll
            for (int c = 0; c < 4; ++c)
                acc[c] += w2 * xb[(4 + c) * HWs + gp];
            msk += w2;
        }
    }
    float hard = (msk >= 0.9999f) ? 1.f : 0.f;
    #pragma unroll
    for (int c = 0; c < 4; ++c)
        ob[(4 + c) * HWs + pix] = acc[c] * hard;
}

// ---------------------------------------------------------------------------
// Kernel 2: conv0 (15x15, pad 7, 8ch -> 4ch) + mask = sigmoid(sum|c0| - 700).
// Implicit GEMM: M = 16 output x-positions, N = 16 (4 real out channels),
// K = 8*15*16 (kx padded 15->16, pad weight = 0) -> 60 x wmma_f32_16x16x32_f16
// per output row of 16 pixels. One workgroup = 16 waves = 16 rows x 16 cols.
// Input tile staged via async global->LDS (interior blocks), f32 in LDS,
// converted to f16 at fragment build (co-executes with WMMA).
// ---------------------------------------------------------------------------
#define C0_ROWS 16
#define C0_TR   (C0_ROWS + 14)   // 30 input rows incl. halo

__global__ __launch_bounds__(512) void conv0_mask_kernel(
    const float* __restrict__ xc,
    const float* __restrict__ w0,
    float* __restrict__ m0)
{
    // weights as B^T: [n][k'], k' = (c*15+ky)*16+kx ; row 4 = shared zero row
    __shared__ _Float16 sW[5 * 1920];
    // input tile halo: [c][row 0..29][col 0..31] (f32)
    __shared__ float sIn[8 * C0_TR * 32];

    const int tid = threadIdx.x;
    const int b   = blockIdx.z;
    const int x0  = blockIdx.x * 16;
    const int y0  = blockIdx.y * C0_ROWS;

    for (int t = tid; t < 5 * 1920; t += 512) {
        int n = t / 1920;
        int k = t - n * 1920;
        int c  = k / 240;
        int r  = k - c * 240;
        int ky = r >> 4;
        int kx = r & 15;
        float v = 0.f;
        if (n < 4 && kx < 15)
            v = w0[((n * 8 + c) * 15 + ky) * 15 + kx];
        sW[t] = (_Float16)v;
    }

    const float* xb = xc + (long long)b * 8 * HWs;
    const bool interior = (blockIdx.x >= 1) && (blockIdx.x <= 30) &&
                          (blockIdx.y >= 1) && (blockIdx.y <= 30);
#if HAVE_ASYNC_LDS
    if (interior) {
        // in-bounds everywhere: straight async byte-mover, no VGPR round trip
        for (int t = tid; t < 8 * C0_TR * 32; t += 512) {
            int c   = t / (C0_TR * 32);
            int r   = (t / 32) % C0_TR;
            int col = t & 31;
            const float* gp = xb + (long long)c * HWs +
                              (long long)(y0 - 7 + r) * WW + (x0 - 7 + col);
            async_copy_b32(gp, &sIn[t]);
        }
        async_wait0();
    } else
#endif
    {
        for (int t = tid; t < 8 * C0_TR * 32; t += 512) {
            int c   = t / (C0_TR * 32);
            int r   = (t / 32) % C0_TR;
            int col = t & 31;
            int gy = y0 - 7 + r;
            int gx = x0 - 7 + col;
            float v = 0.f;
            if (gy >= 0 && gy < HH && gx >= 0 && gx < WW)
                v = xb[(long long)c * HWs + (long long)gy * WW + gx];
            sIn[t] = v;
        }
    }
    __syncthreads();

    const int wave = tid >> 5;        // output row within tile
    const int lane = tid & 31;
    const int m    = lane & 15;       // A row / B column (out channel)
    const int hi   = lane >> 4;       // half-wave selector

    v8f acc = {0.f, 0.f, 0.f, 0.f, 0.f, 0.f, 0.f, 0.f};
    const _Float16* pB = &sW[(m < 4 ? m : 4) * 1920 + hi * 8];

    int c0 = 0, ky0 = 0;
    for (int kb = 0; kb < 60; ++kb) {
        int c1 = c0, ky1 = ky0 + 1;
        if (ky1 == 15) { ky1 = 0; ++c1; }
        const float* pA0 = &sIn[(c0 * C0_TR + (wave + ky0)) * 32 + m + hi * 8];
        const float* pA1 = &sIn[(c1 * C0_TR + (wave + ky1)) * 32 + m + hi * 8];
        v16h a, bf;
        #pragma unroll
        for (int e = 0; e < 8; ++e) {
            a[e]     = (_Float16)pA0[e];
            a[e + 8] = (_Float16)pA1[e];
        }
        #pragma unroll
        for (int e = 0; e < 16; ++e) bf[e] = pB[kb * 32 + (e < 8 ? e : e + 8)];
        acc = __builtin_amdgcn_wmma_f32_16x16x32_f16(
                  false, a, false, bf, (short)0, acc, false, false);
        ky0 += 2;
        if (ky0 >= 15) { ky0 -= 15; ++c0; }
    }

    // acc[v] on lane L: output pixel x0 + v + 8*hi, channel (L&15) (0 for n>=4)
    float* mrow = m0 + ((long long)b * HH + (y0 + wave)) * WW + x0;
    #pragma unroll
    for (int v = 0; v < 8; ++v) {
        float s = half_wave_sum(fabsf(acc[v]));
        if (m == 0) {
            float mask = 1.f / (1.f + __expf(700.f - s));
            mrow[v + 8 * hi] = mask;
        }
    }
}

// ---------------------------------------------------------------------------
// Kernel 3: 2x2 max-pool of the mask (B,1,H,W) -> (B,1,H/2,W/2)
// ---------------------------------------------------------------------------
__global__ void maxpool_kernel(const float* __restrict__ in,
                               float* __restrict__ out,
                               int Hin, int total)
{
    int idx = blockIdx.x * blockDim.x + threadIdx.x;
    if (idx >= total) return;
    int Hout = Hin >> 1;
    int x = idx % Hout;
    int t = idx / Hout;
    int y = t % Hout;
    int b = t / Hout;
    const float* ib = in + (long long)b * Hin * Hin;
    float a0 = ib[(2 * y) * Hin + 2 * x];
    float a1 = ib[(2 * y) * Hin + 2 * x + 1];
    float a2 = ib[(2 * y + 1) * Hin + 2 * x];
    float a3 = ib[(2 * y + 1) * Hin + 2 * x + 1];
    out[idx] = fmaxf(fmaxf(a0, a1), fmaxf(a2, a3));
}

// ---------------------------------------------------------------------------
// Kernel 4: down(): reflect-pad (0,1)x(0,1) then 3x3 stride-2 conv, 4->4 ch.
// ---------------------------------------------------------------------------
__global__ void down_kernel(const float* __restrict__ in,
                            long long inBS, long long inCS,
                            const float* __restrict__ w3,
                            float* __restrict__ out,
                            int Hin, int total)
{
    int idx = blockIdx.x * blockDim.x + threadIdx.x;
    if (idx >= total) return;
    int Hout = Hin >> 1;
    int x = idx % Hout;
    int t = idx / Hout;
    int y = t % Hout;
    t /= Hout;
    int o = t & 3;
    int b = t >> 2;

    const float* ib = in + (long long)b * inBS;
    float s = 0.f;
    #pragma unroll
    for (int c = 0; c < 4; ++c) {
        #pragma unroll
        for (int ky = 0; ky < 3; ++ky) {
            int r = 2 * y + ky; if (r >= Hin) r = Hin - 2;   // reflect
            #pragma unroll
            for (int kx = 0; kx < 3; ++kx) {
                int cc = 2 * x + kx; if (cc >= Hin) cc = Hin - 2;
                s += w3[((o * 4 + c) * 3 + ky) * 3 + kx] *
                     ib[c * inCS + (long long)r * Hin + cc];
            }
        }
    }
    out[idx] = s;
}

// ---------------------------------------------------------------------------
// Kernel 5: pyramid top: ls64 = g3_ref*m3 + g3_alt*(1-m3)
// ---------------------------------------------------------------------------
__global__ void blend_top_kernel(const float* __restrict__ gr,
                                 const float* __restrict__ ga,
                                 const float* __restrict__ m3,
                                 float* __restrict__ out,
                                 int Hc, int total)
{
    int idx = blockIdx.x * blockDim.x + threadIdx.x;
    if (idx >= total) return;
    int x = idx % Hc;
    int t = idx / Hc;
    int y = t % Hc;
    t /= Hc;
    int b = t >> 2;
    float mm = m3[((long long)b * Hc + y) * Hc + x];
    out[idx] = gr[idx] * mm + ga[idx] * (1.f - mm);
}

// ---------------------------------------------------------------------------
// Kernel 6: fused collapse level:
//  out = up(prev,w) + (g_ref - up(gc_ref,w))*m + (g_alt - up(gc_alt,w))*(1-m)
//      = g_ref*m + g_alt*(1-m) + sum_c w[c,o,p,q]*(prev[c]-m*gcr[c]-(1-m)*gca[c])
// ---------------------------------------------------------------------------
__global__ void up_blend_kernel(const float* __restrict__ prev,   // (B,4,Hc,Hc)
                                const float* __restrict__ gcr,    // (B,4,Hc,Hc)
                                const float* __restrict__ gca,    // (B,4,Hc,Hc)
                                const float* __restrict__ gfr,    // fine ref
                                const float* __restrict__ gfa,    // fine alt
                                long long gfBS, long long gfCS,
                                const float* __restrict__ mf,     // (B,1,Hf,Hf)
                                const float* __restrict__ upw,    // (c,o,2,2)
                                float* __restrict__ out,          // (B,4,Hf,Hf)
                                int Hf, int total)
{
    int idx = blockIdx.x * blockDim.x + threadIdx.x;
    if (idx >= total) return;
    int Hc = Hf >> 1;
    int x = idx % Hf;
    int t = idx / Hf;
    int y = t % Hf;
    t /= Hf;
    int o = t & 3;
    int b = t >> 2;

    int yc = y >> 1, xc2 = x >> 1, p = y & 1, q = x & 1;
    float mm = mf[((long long)b * Hf + y) * Hf + x];

    float s = gfr[(long long)b * gfBS + o * gfCS + (long long)y * Hf + x] * mm
            + gfa[(long long)b * gfBS + o * gfCS + (long long)y * Hf + x] * (1.f - mm);

    long long cbase = ((long long)b * 4) * Hc * Hc + (long long)yc * Hc + xc2;
    #pragma unroll
    for (int c = 0; c < 4; ++c) {
        long long ci = cbase + (long long)c * Hc * Hc;
        float tt = prev[ci] - mm * gcr[ci] - (1.f - mm) * gca[ci];
        s += upw[((c * 4 + o) * 2 + p) * 2 + q] * tt;
    }
    out[idx] = s;
}

// ---------------------------------------------------------------------------
// Host launcher
// ---------------------------------------------------------------------------
extern "C" void kernel_launch(void* const* d_in, const int* in_sizes, int n_in,
                              void* d_out, int out_size, void* d_ws, size_t ws_size,
                              hipStream_t stream)
{
    const float* x  = (const float*)d_in[0];
    const float* w0 = (const float*)d_in[1];
    const float* w1 = (const float*)d_in[2];
    const float* w2 = (const float*)d_in[3];
    const float* w3 = (const float*)d_in[4];
    const float* u1 = (const float*)d_in[5];
    const float* u2 = (const float*)d_in[6];
    const float* u3 = (const float*)d_in[7];
    float* out = (float*)d_out;
    float* ws  = (float*)d_ws;

    // workspace layout (floats), ~111 MB total
    long long off = 0;
    float* xc  = ws + off; off += 8 * 8 * HWs;            // (8,8,512,512)
    float* m0  = ws + off; off += 8 * HWs;                // (8,1,512,512)
    float* m1  = ws + off; off += 8LL * 256 * 256;
    float* m2  = ws + off; off += 8LL * 128 * 128;
    float* m3  = ws + off; off += 8LL * 64 * 64;
    float* g1r = ws + off; off += 8LL * 4 * 256 * 256;
    float* g1a = ws + off; off += 8LL * 4 * 256 * 256;
    float* g2r = ws + off; off += 8LL * 4 * 128 * 128;
    float* g2a = ws + off; off += 8LL * 4 * 128 * 128;
    float* g3r = ws + off; off += 8LL * 4 * 64 * 64;
    float* g3a = ws + off; off += 8LL * 4 * 64 * 64;
    float* ls64  = ws + off; off += 8LL * 4 * 64 * 64;
    float* ls128 = ws + off; off += 8LL * 4 * 128 * 128;
    float* ls256 = ws + off; off += 8LL * 4 * 256 * 256;

    const int BLK = 256;
    #define NB(n) (int)(((n) + BLK - 1) / BLK)

    // 1) warp + concat -> xc
    {
        int total = (int)(8 * HWs);
        warp_concat_kernel<<<NB(total), BLK, 0, stream>>>(x, xc, total);
    }

    // 2) conv0 + mask -> m0   (WMMA, async-LDS staging)
    {
        dim3 grid(WW / 16, HH / C0_ROWS, 8);
        conv0_mask_kernel<<<grid, 512, 0, stream>>>(xc, w0, m0);
    }

    // 3) mask pyramid
    {
        int t1 = 8 * 256 * 256, t2 = 8 * 128 * 128, t3 = 8 * 64 * 64;
        maxpool_kernel<<<NB(t1), BLK, 0, stream>>>(m0, m1, 512, t1);
        maxpool_kernel<<<NB(t2), BLK, 0, stream>>>(m1, m2, 256, t2);
        maxpool_kernel<<<NB(t3), BLK, 0, stream>>>(m2, m3, 128, t3);
    }

    // 4) Gaussian pyramids (ref = xc[:, :4], alt = xc[:, 4:8])
    {
        int t1 = 8 * 4 * 256 * 256;
        down_kernel<<<NB(t1), BLK, 0, stream>>>(xc,            8 * HWs, HWs, w1, g1r, 512, t1);
        down_kernel<<<NB(t1), BLK, 0, stream>>>(xc + 4 * HWs,  8 * HWs, HWs, w1, g1a, 512, t1);
        int t2 = 8 * 4 * 128 * 128;
        down_kernel<<<NB(t2), BLK, 0, stream>>>(g1r, 4LL * 256 * 256, 256LL * 256, w2, g2r, 256, t2);
        down_kernel<<<NB(t2), BLK, 0, stream>>>(g1a, 4LL * 256 * 256, 256LL * 256, w2, g2a, 256, t2);
        int t3 = 8 * 4 * 64 * 64;
        down_kernel<<<NB(t3), BLK, 0, stream>>>(g2r, 4LL * 128 * 128, 128LL * 128, w3, g3r, 128, t3);
        down_kernel<<<NB(t3), BLK, 0, stream>>>(g2a, 4LL * 128 * 128, 128LL * 128, w3, g3a, 128, t3);
    }

    // 5) blend top level (64)
    {
        int t = 8 * 4 * 64 * 64;
        blend_top_kernel<<<NB(t), BLK, 0, stream>>>(g3r, g3a, m3, ls64, 64, t);
    }

    // 6) collapse: 64->128 (up1_w, m2), 128->256 (up2_w, m1), 256->512 (up3_w, m0)
    {
        int t128 = 8 * 4 * 128 * 128;
        up_blend_kernel<<<NB(t128), BLK, 0, stream>>>(
            ls64, g3r, g3a, g2r, g2a, 4LL * 128 * 128, 128LL * 128,
            m2, u1, ls128, 128, t128);
        int t256 = 8 * 4 * 256 * 256;
        up_blend_kernel<<<NB(t256), BLK, 0, stream>>>(
            ls128, g2r, g2a, g1r, g1a, 4LL * 256 * 256, 256LL * 256,
            m1, u2, ls256, 256, t256);
        int t512 = (int)(8 * 4 * HWs);
        up_blend_kernel<<<NB(t512), BLK, 0, stream>>>(
            ls256, g1r, g1a, xc, xc + 4 * HWs, 8 * HWs, HWs,
            m0, u3, out, 512, t512);
    }
    #undef NB
}
